// SimpleEdgeCNN_29403346108562
// MI455X (gfx1250) — matrix operand order repaired and checked by reading, not compiled, not run on hardware
//
#include <hip/hip_runtime.h>
#include <hip/hip_bf16.h>

typedef __attribute__((ext_vector_type(16))) _Float16 v16h;
typedef __attribute__((ext_vector_type(8)))  float    v8f;
typedef __attribute__((ext_vector_type(4)))  float    v4f;

#define IN_C   32
#define HID    64
#define OUT_C  32
#define TILE   16
#define WPB    8      // waves per block (256 threads, wave32)
#define HSTR   68     // padded LDS row stride (floats) for H tile

// float atomic max via int-max / uint-min trick (valid with -inf init)
__device__ __forceinline__ void atomic_max_f32(float* addr, float val) {
    if (val > 0.0f) {
        atomicMax((int*)addr, __float_as_int(val));
    } else {
        atomicMin((unsigned int*)addr, __float_as_uint(val));
    }
}

__global__ void __launch_bounds__(256) edgeconv_init(float* out, int n) {
    int i = blockIdx.x * blockDim.x + threadIdx.x;
    if (i < n) out[i] = -__builtin_huge_valf();
}

__global__ void __launch_bounds__(256) edgeconv_finalize(float* out, int n) {
    int i = blockIdx.x * blockDim.x + threadIdx.x;
    if (i < n) {
        float v = out[i];
        out[i] = (v == -__builtin_huge_valf()) ? 0.0f : v;
    }
}

__global__ void __launch_bounds__(256) edgeconv_wmma(
    const float* __restrict__ x,
    const long long* __restrict__ esrc,
    const long long* __restrict__ edst,
    const float* __restrict__ W1, const float* __restrict__ b1,
    const float* __restrict__ W2, const float* __restrict__ b2,
    float* __restrict__ out,
    int E, int nTiles, int tileBlocks)
{
    __shared__ __align__(16) float Hlds[WPB * TILE * HSTR];
    __shared__ int Dlds[WPB * TILE];

    const int lane  = threadIdx.x & 31;
    const int wid   = threadIdx.x >> 5;
    const int col15 = lane & 15;
    const int hi    = lane >> 4;

    // ---- loop-invariant weight fragments in WMMA B layout ----
    // B (32x16 f16): lane holds col = lane&15; e<8 -> K=hi*8+e ; e>=8 -> K=16+hi*8+(e-8)
    v16h bw1[2][4];
    #pragma unroll
    for (int kc = 0; kc < 2; ++kc)
      #pragma unroll
      for (int n = 0; n < 4; ++n) {
        v16h f;
        #pragma unroll
        for (int e = 0; e < 16; ++e)
          f[e] = (_Float16)W1[(kc*32 + hi*8 + (e & 7) + (e >> 3)*16) * HID + n*16 + col15];
        bw1[kc][n] = f;
      }
    v16h bw2[2][2];
    #pragma unroll
    for (int kc = 0; kc < 2; ++kc)
      #pragma unroll
      for (int n = 0; n < 2; ++n) {
        v16h f;
        #pragma unroll
        for (int e = 0; e < 16; ++e)
          f[e] = (_Float16)W2[(kc*32 + hi*8 + (e & 7) + (e >> 3)*16) * OUT_C + n*16 + col15];
        bw2[kc][n] = f;
      }

    float* Hw = Hlds + wid * (TILE * HSTR);
    int*   Dw = Dlds + wid * TILE;

    for (int tb = blockIdx.x; tb < tileBlocks; tb += gridDim.x) {
        const int t = tb * WPB + wid;
        const bool active = (t < nTiles);

        if (active) {
            const int e0 = t * TILE;
            int er = e0 + col15; if (er > E - 1) er = E - 1;   // idempotent under max
            const long long sj = esrc[er];
            const long long di = edst[er];
            if (lane < 16) Dw[lane] = (int)di;

            const float* xi = x + (size_t)di * IN_C;
            const float* xj = x + (size_t)sj * IN_C;

            // A layout (16x32 f16): row = lane&15; e<8 -> K=hi*8+e ; e>=8 -> K=16+hi*8+(e-8)
            v4f i0 = *(const v4f*)(xi + hi*8);
            v4f i1 = *(const v4f*)(xi + hi*8 + 4);
            v4f i2 = *(const v4f*)(xi + 16 + hi*8);
            v4f i3 = *(const v4f*)(xi + 20 + hi*8);
            v4f j0 = *(const v4f*)(xj + hi*8);
            v4f j1 = *(const v4f*)(xj + hi*8 + 4);
            v4f j2 = *(const v4f*)(xj + 16 + hi*8);
            v4f j3 = *(const v4f*)(xj + 20 + hi*8);

            v16h a0, a1;   // a0 = x_i (m cols 0..31), a1 = x_j - x_i (m cols 32..63)
            #pragma unroll
            for (int q = 0; q < 4; ++q) {
                a0[q]    = (_Float16)i0[q];
                a0[4+q]  = (_Float16)i1[q];
                a0[8+q]  = (_Float16)i2[q];
                a0[12+q] = (_Float16)i3[q];
                a1[q]    = (_Float16)(j0[q] - i0[q]);
                a1[4+q]  = (_Float16)(j1[q] - i1[q]);
                a1[8+q]  = (_Float16)(j2[q] - i2[q]);
                a1[12+q] = (_Float16)(j3[q] - i3[q]);
            }

            // layer 1: H = relu(m @ W1 + b1), 4 N-tiles x 2 K-chunks = 8 WMMAs
            #pragma unroll
            for (int n = 0; n < 4; ++n) {
                float bv = b1[n*16 + col15];
                v8f acc;
                #pragma unroll
                for (int q = 0; q < 8; ++q) acc[q] = bv;
                acc = __builtin_amdgcn_wmma_f32_16x16x32_f16(false, a0, false, bw1[0][n],
                                                             (short)0, acc, false, false);
                acc = __builtin_amdgcn_wmma_f32_16x16x32_f16(false, a1, false, bw1[1][n],
                                                             (short)0, acc, false, false);
                // C layout: row = v + 8*hi, col = 16n + col15 ; ReLU + stage to LDS
                #pragma unroll
                for (int v = 0; v < 8; ++v) {
                    float h = acc[v] > 0.0f ? acc[v] : 0.0f;
                    Hw[(v + 8*hi) * HSTR + n*16 + col15] = h;
                }
            }
        }
        __syncthreads();
        if (active) {
            // re-read H in A layout (transpose through LDS), convert to f16
            const float* hr = Hw + col15 * HSTR;
            v4f h0 = *(const v4f*)(hr + hi*8);
            v4f h1 = *(const v4f*)(hr + hi*8 + 4);
            v4f h2 = *(const v4f*)(hr + 16 + hi*8);
            v4f h3 = *(const v4f*)(hr + 20 + hi*8);
            v4f h4 = *(const v4f*)(hr + 32 + hi*8);
            v4f h5 = *(const v4f*)(hr + 36 + hi*8);
            v4f h6 = *(const v4f*)(hr + 48 + hi*8);
            v4f h7 = *(const v4f*)(hr + 52 + hi*8);
            v16h p0, p1;
            #pragma unroll
            for (int q = 0; q < 4; ++q) {
                p0[q]    = (_Float16)h0[q]; p0[4+q]  = (_Float16)h1[q];
                p0[8+q]  = (_Float16)h2[q]; p0[12+q] = (_Float16)h3[q];
                p1[q]    = (_Float16)h4[q]; p1[4+q]  = (_Float16)h5[q];
                p1[8+q]  = (_Float16)h6[q]; p1[12+q] = (_Float16)h7[q];
            }
            // layer 2: o = H @ W2 + b2, 2 N-tiles x 2 K-chunks = 4 WMMAs
            #pragma unroll
            for (int n = 0; n < 2; ++n) {
                float bv = b2[n*16 + col15];
                v8f oc;
                #pragma unroll
                for (int q = 0; q < 8; ++q) oc[q] = bv;
                oc = __builtin_amdgcn_wmma_f32_16x16x32_f16(false, p0, false, bw2[0][n],
                                                            (short)0, oc, false, false);
                oc = __builtin_amdgcn_wmma_f32_16x16x32_f16(false, p1, false, bw2[1][n],
                                                            (short)0, oc, false, false);
                // segment-max scatter: row = v + 8*hi is the edge within the tile
                #pragma unroll
                for (int v = 0; v < 8; ++v) {
                    int d = Dw[v + 8*hi];
                    atomic_max_f32(out + (size_t)d * OUT_C + n*16 + col15, oc[v]);
                }
            }
        }
        __syncthreads();
    }
}

extern "C" void kernel_launch(void* const* d_in, const int* in_sizes, int n_in,
                              void* d_out, int out_size, void* d_ws, size_t ws_size,
                              hipStream_t stream) {
    const float*     x  = (const float*)d_in[0];
    const long long* ei = (const long long*)d_in[1];
    const float*     W1 = (const float*)d_in[2];
    const float*     b1 = (const float*)d_in[3];
    const float*     W2 = (const float*)d_in[4];
    const float*     b2 = (const float*)d_in[5];
    float* out = (float*)d_out;

    const int E = in_sizes[1] / 2;
    const long long* esrc = ei;       // edge_index[0] = j (neighbor)
    const long long* edst = ei + E;   // edge_index[1] = i (center)

    edgeconv_init<<<(out_size + 255) / 256, 256, 0, stream>>>(out, out_size);

    const int nTiles     = (E + TILE - 1) / TILE;
    const int tileBlocks = (nTiles + WPB - 1) / WPB;
    const int grid       = tileBlocks < 1536 ? tileBlocks : 1536;
    edgeconv_wmma<<<grid, 256, 0, stream>>>(x, esrc, edst, W1, b1, W2, b2, out,
                                            E, nTiles, tileBlocks);

    edgeconv_finalize<<<(out_size + 255) / 256, 256, 0, stream>>>(out, out_size);
}
